// Attention_53231824666818
// MI455X (gfx1250) — matrix operand order repaired and checked by reading, not compiled, not run on hardware
//
#include <hip/hip_runtime.h>
#include <math.h>

// ---------------- problem constants ----------------
#define S_   2048
#define HID_ 2048
#define NH_  16
#define NKV_ 2
#define HD_  128
#define G_   (NH_ / NKV_)

typedef __bf16 bf16_t;
typedef __attribute__((ext_vector_type(16))) __bf16 v16bf;
typedef __attribute__((ext_vector_type(8)))  __bf16 v8bf;
typedef __attribute__((ext_vector_type(8)))  float  v8f;

static __device__ inline v8f wmma_bf16(v16bf a, v16bf b, v8f c) {
    // D = A(16x32 bf16) * B(32x16 bf16) + C(16x16 f32)
    return __builtin_amdgcn_wmma_f32_16x16x32_bf16(
        /*neg_a=*/false, a, /*neg_b=*/false, b,
        /*c_mod=*/(short)0, c, /*reuse_a=*/false, /*reuse_b=*/false);
}

// ---------------- f32 -> bf16 convert ----------------
__global__ void cvt_bf16_kernel(const float* __restrict__ in,
                                bf16_t* __restrict__ out, int n) {
    int i = blockIdx.x * blockDim.x + threadIdx.x;
    if (i < n) out[i] = (bf16_t)in[i];
}

// ---------------- bf16 GEMM: C[f32, MxN] = A[bf16, MxK] @ B[bf16, KxN] ----
// 1 wave per block; each wave computes a 64x64 tile (4x4 grid of 16x16 WMMA
// accumulators).  Per 32-wide k-step: 8 b128 A loads + 64 2-byte B loads -> 16 back-to-back WMMAs
// (A and B fragments each reused 4x from registers).
// Requires M%64==0, N%64==0, K%32==0.
__global__ __launch_bounds__(32)
void gemm_bf16_kernel(const bf16_t* __restrict__ A,
                      const bf16_t* __restrict__ B,
                      float* __restrict__ C, int M, int N, int K) {
    const int n0   = blockIdx.x * 64;
    const int m0   = blockIdx.y * 64;
    const int lane = threadIdx.x;
    const int ll   = lane & 15;
    const int lh   = lane >> 4;   // 0 or 1

    // A-operand layout (16-bit A 16x32): lane<16 -> row +ll, K = {0..7,16..23}
    //                                    lane>=16 -> row +ll, K = {8..15,24..31}
    const bf16_t* arow0 = A + (size_t)(m0 + ll) * K + lh * 8;

    v8f acc[4][4];
#pragma unroll
    for (int i = 0; i < 4; ++i)
#pragma unroll
        for (int j = 0; j < 4; ++j)
#pragma unroll
            for (int r = 0; r < 8; ++r) acc[i][j][r] = 0.0f;

    for (int k0 = 0; k0 < K; k0 += 32) {
        // ---- load phase: all fragments for this k-step ----
        v16bf a[4], b[4];
#pragma unroll
        for (int i = 0; i < 4; ++i) {
            const bf16_t* ap = arow0 + (size_t)(16 * i) * K + k0;
            v8bf lo = *(const v8bf*)ap;
            v8bf hi = *(const v8bf*)(ap + 16);
#pragma unroll
            for (int e = 0; e < 8; ++e) { a[i][e] = lo[e]; a[i][e + 8] = hi[e]; }
        }
        // B-operand layout (16-bit B 32x16): lane -> col +ll, K = {0..7,16..23} / {8..15,24..31}
        const bf16_t* brow = B + (size_t)(k0 + lane) * N + n0;
        const bf16_t* bcol = B + (size_t)k0 * N + n0 + ll;
#pragma unroll
        for (int j = 0; j < 4; ++j)
#pragma unroll
            for (int e = 0; e < 8; ++e) {
                b[j][e]     = bcol[(size_t)(8 * lh + e) * N + 16 * j];
                b[j][e + 8] = bcol[(size_t)(16 + 8 * lh + e) * N + 16 * j];
            }

        if (k0 + 32 < K) {
            __builtin_prefetch(arow0 + k0 + 32, 0, 3);
            __builtin_prefetch(brow + (size_t)32 * N, 0, 3);
        }

        // ---- math phase: 16 WMMAs ----
#pragma unroll
        for (int i = 0; i < 4; ++i)
#pragma unroll
            for (int j = 0; j < 4; ++j)
                acc[i][j] = wmma_bf16(a[i], b[j], acc[i][j]);
    }

    // C/D layout: VGPR r, lanes0-15 -> (M=r, N=ll); lanes16-31 -> (M=8+r, N=ll)
#pragma unroll
    for (int i = 0; i < 4; ++i)
#pragma unroll
        for (int j = 0; j < 4; ++j)
#pragma unroll
            for (int r = 0; r < 8; ++r)
                C[(size_t)(m0 + 16 * i + r + 8 * lh) * N + n0 + 16 * j + ll] =
                    acc[i][j][r];
}

// ---------------- fused RMSNorm + RoPE ----------------
// grid (S, nheads), block HD.  transposed=0 -> out[h][s][d]; =1 -> out[h][d][s]
__global__ __launch_bounds__(HD_)
void rope_norm_kernel(const float* __restrict__ proj,
                      const float* __restrict__ w,
                      const int* __restrict__ pos_ids,
                      bf16_t* __restrict__ out, int nheads, int transposed) {
    const int s = blockIdx.x, h = blockIdx.y, d = threadIdx.x;
    __shared__ float red[HD_];
    __shared__ float xn[HD_];

    float x = proj[(size_t)s * nheads * HD_ + h * HD_ + d];
    red[d] = x * x;
    __syncthreads();
#pragma unroll
    for (int o2 = HD_ / 2; o2 > 0; o2 >>= 1) {
        if (d < o2) red[d] += red[d + o2];
        __syncthreads();
    }
    float rms = rsqrtf(red[0] * (1.0f / HD_) + 1e-6f);
    float v = x * rms * w[d];
    xn[d] = v;
    __syncthreads();
    float other = xn[d ^ 64];
    float rh = (d < 64) ? -other : other;

    float p = (float)pos_ids[s];
    // inv_freq = THETA^(-(d%64)/64) = exp(-(d%64) * ln(1e6)/64)
    float fr = p * __expf(-(float)(d & 63) * 0.21586735246819178f);
    float outv = v * __cosf(fr) + rh * __sinf(fr);

    size_t oi = transposed ? ((size_t)(h * HD_ + d) * S_ + s)
                           : ((size_t)h * S_ * HD_ + (size_t)s * HD_ + d);
    out[oi] = (bf16_t)outv;
}

// ---------------- V repack: [s, kv*HD] f32 -> [kv, s, HD] bf16 -------------
__global__ void vpack_kernel(const float* __restrict__ vproj,
                             bf16_t* __restrict__ Vb) {
    int i = blockIdx.x * blockDim.x + threadIdx.x;
    if (i >= S_ * NKV_ * HD_) return;
    int s = i >> 8;           // / (NKV_*HD_)
    int r = i & 255;
    int kh = r >> 7;
    int d = r & 127;
    Vb[(size_t)kh * S_ * HD_ + (size_t)s * HD_ + d] = (bf16_t)vproj[i];
}

// ---------------- flash-style causal GQA attention ----------------
// grid (S/16, NH), 1 wave per block. Q[h][s][d], Kt[kv][d][s], V[kv][s][d].
__global__ __launch_bounds__(32)
void attn_kernel(const bf16_t* __restrict__ Qb,
                 const bf16_t* __restrict__ Ktb,
                 const bf16_t* __restrict__ Vb,
                 bf16_t* __restrict__ attnb) {
    const int q0   = blockIdx.x * 16;
    const int h    = blockIdx.y;
    const int kh   = h / G_;
    const int lane = threadIdx.x;
    const int ll   = lane & 15;
    const int lh   = lane >> 4;

    __shared__ float pt[16 * 32];   // P staging (C-layout -> A-layout relay)

    // Load Q tile (16x128) as 4 A-operands of 16x32
    v16bf qa[4];
    const bf16_t* qrow = Qb + (size_t)h * S_ * HD_ + (size_t)(q0 + ll) * HD_;
#pragma unroll
    for (int c = 0; c < 4; ++c) {
        const bf16_t* p = qrow + 32 * c + lh * 8;
        v8bf lo = *(const v8bf*)p;
        v8bf hi = *(const v8bf*)(p + 16);
#pragma unroll
        for (int i = 0; i < 8; ++i) { qa[c][i] = lo[i]; qa[c][i + 8] = hi[i]; }
    }

    v8f o[8];
    float mrow[8], lrow[8];
#pragma unroll
    for (int t = 0; t < 8; ++t)
#pragma unroll
        for (int r = 0; r < 8; ++r) o[t][r] = 0.0f;
#pragma unroll
    for (int r = 0; r < 8; ++r) { mrow[r] = -1e30f; lrow[r] = 0.0f; }

    const float scale = 0.08838834764831845f;   // 1/sqrt(128)
    const int kend = q0 + 16;

    for (int kt0 = 0; kt0 < kend; kt0 += 32) {
        // scores for 32 keys: two 16x16 C tiles, K-dim chained over HD
        v8f sc0, sc1;
#pragma unroll
        for (int r = 0; r < 8; ++r) { sc0[r] = 0.0f; sc1[r] = 0.0f; }
#pragma unroll
        for (int c = 0; c < 4; ++c) {
            const bf16_t* kb = Ktb + ((size_t)kh * HD_ + 32 * c) * S_ + kt0 + ll;
            v16bf b0, b1;
#pragma unroll
            for (int e = 0; e < 8; ++e) {
                b0[e]     = kb[(size_t)(8 * lh + e) * S_];
                b0[e + 8] = kb[(size_t)(16 + 8 * lh + e) * S_];
                b1[e]     = kb[(size_t)(8 * lh + e) * S_ + 16];
                b1[e + 8] = kb[(size_t)(16 + 8 * lh + e) * S_ + 16];
            }
            sc0 = wmma_bf16(qa[c], b0, sc0);
            sc1 = wmma_bf16(qa[c], b1, sc1);
        }

        // scale + causal mask + online softmax (per-row stats across 16-lane half)
#pragma unroll
        for (int r = 0; r < 8; ++r) {
            int qg = q0 + r + 8 * lh;
            float s0v = sc0[r] * scale;
            float s1v = sc1[r] * scale;
            if (kt0 + ll > qg)      s0v = -3e38f;
            if (kt0 + 16 + ll > qg) s1v = -3e38f;

            float mx = fmaxf(s0v, s1v);
            mx = fmaxf(mx, __shfl_xor(mx, 1));
            mx = fmaxf(mx, __shfl_xor(mx, 2));
            mx = fmaxf(mx, __shfl_xor(mx, 4));
            mx = fmaxf(mx, __shfl_xor(mx, 8));
            float mnew = fmaxf(mrow[r], mx);
            float alpha = __expf(mrow[r] - mnew);
            mrow[r] = mnew;
            float p0 = __expf(s0v - mnew);
            float p1 = __expf(s1v - mnew);
            sc0[r] = p0; sc1[r] = p1;
            float rs = p0 + p1;
            rs += __shfl_xor(rs, 1);
            rs += __shfl_xor(rs, 2);
            rs += __shfl_xor(rs, 4);
            rs += __shfl_xor(rs, 8);
            lrow[r] = lrow[r] * alpha + rs;
#pragma unroll
            for (int t = 0; t < 8; ++t) o[t][r] *= alpha;
        }

        // relay P (C-layout f32) -> A-layout bf16 through LDS
#pragma unroll
        for (int r = 0; r < 8; ++r) {
            pt[(r + 8 * lh) * 32 + ll]      = sc0[r];
            pt[(r + 8 * lh) * 32 + 16 + ll] = sc1[r];
        }
        __syncthreads();
        v16bf pa;
        {
            const float* prow = pt + ll * 32 + lh * 8;
#pragma unroll
            for (int i = 0; i < 8; ++i) {
                pa[i]     = (bf16_t)prow[i];
                pa[i + 8] = (bf16_t)prow[16 + i];
            }
        }
        __syncthreads();

        // O += P(16x32) @ V(32x128): lane holds V column d = 16t+ll, keys in table order
        const bf16_t* vrow = Vb + (size_t)kh * S_ * HD_ + (size_t)kt0 * HD_ + ll;
#pragma unroll
        for (int t = 0; t < 8; ++t) {
            v16bf vb;
#pragma unroll
            for (int e = 0; e < 8; ++e) {
                vb[e]     = vrow[(size_t)(8 * lh + e) * HD_ + 16 * t];
                vb[e + 8] = vrow[(size_t)(16 + 8 * lh + e) * HD_ + 16 * t];
            }
            o[t] = wmma_bf16(pa, vb, o[t]);
        }
    }

    // epilogue: normalize and store [s, NH*HD] bf16
#pragma unroll
    for (int t = 0; t < 8; ++t)
#pragma unroll
        for (int r = 0; r < 8; ++r) {
            float val = o[t][r] / lrow[r];
            int row = q0 + r + 8 * lh;
            attnb[(size_t)row * (NH_ * HD_) + h * HD_ + 16 * t + ll] = (bf16_t)val;
        }
}

// ---------------- host launcher ----------------
extern "C" void kernel_launch(void* const* d_in, const int* in_sizes, int n_in,
                              void* d_out, int out_size, void* d_ws, size_t ws_size,
                              hipStream_t stream) {
    (void)in_sizes; (void)n_in; (void)out_size; (void)ws_size;
    const float* hidden = (const float*)d_in[0];
    const float* Wq     = (const float*)d_in[1];
    const float* Wk     = (const float*)d_in[2];
    const float* Wv     = (const float*)d_in[3];
    const float* Wo     = (const float*)d_in[4];
    const float* qnw    = (const float*)d_in[5];
    const float* knw    = (const float*)d_in[6];
    const int*   pos    = (const int*)d_in[7];

    char* w = (char*)d_ws;
    size_t off = 0;
    auto take = [&](size_t bytes) -> char* {
        char* p = w + off;
        off += (bytes + 255) & ~(size_t)255;
        return p;
    };

    bf16_t* hb    = (bf16_t*)take((size_t)S_ * HID_ * 2);
    bf16_t* wqb   = (bf16_t*)take((size_t)HID_ * NH_ * HD_ * 2);
    bf16_t* wkb   = (bf16_t*)take((size_t)HID_ * NKV_ * HD_ * 2);
    bf16_t* wvb   = (bf16_t*)take((size_t)HID_ * NKV_ * HD_ * 2);
    bf16_t* wob   = (bf16_t*)take((size_t)NH_ * HD_ * HID_ * 2);
    float*  qproj = (float*)take((size_t)S_ * NH_ * HD_ * 4);
    float*  kproj = (float*)take((size_t)S_ * NKV_ * HD_ * 4);
    float*  vproj = (float*)take((size_t)S_ * NKV_ * HD_ * 4);
    bf16_t* Qb    = (bf16_t*)take((size_t)NH_ * S_ * HD_ * 2);
    bf16_t* Ktb   = (bf16_t*)take((size_t)NKV_ * HD_ * S_ * 2);
    bf16_t* Vb    = (bf16_t*)take((size_t)NKV_ * S_ * HD_ * 2);
    bf16_t* attnb = (bf16_t*)take((size_t)S_ * NH_ * HD_ * 2);

    auto cvt = [&](const float* src, bf16_t* dst, int n) {
        cvt_bf16_kernel<<<(n + 255) / 256, 256, 0, stream>>>(src, dst, n);
    };
    cvt(hidden, hb,  S_ * HID_);
    cvt(Wq,     wqb, HID_ * NH_ * HD_);
    cvt(Wk,     wkb, HID_ * NKV_ * HD_);
    cvt(Wv,     wvb, HID_ * NKV_ * HD_);
    cvt(Wo,     wob, NH_ * HD_ * HID_);

    // projections (64x64 tiles per wave)
    gemm_bf16_kernel<<<dim3((NH_ * HD_) / 64, S_ / 64), 32, 0, stream>>>(
        hb, wqb, qproj, S_, NH_ * HD_, HID_);
    gemm_bf16_kernel<<<dim3((NKV_ * HD_) / 64, S_ / 64), 32, 0, stream>>>(
        hb, wkb, kproj, S_, NKV_ * HD_, HID_);
    gemm_bf16_kernel<<<dim3((NKV_ * HD_) / 64, S_ / 64), 32, 0, stream>>>(
        hb, wvb, vproj, S_, NKV_ * HD_, HID_);

    // norm + rope (+ transposes) and V repack
    rope_norm_kernel<<<dim3(S_, NH_),  HD_, 0, stream>>>(qproj, qnw, pos, Qb,  NH_,  0);
    rope_norm_kernel<<<dim3(S_, NKV_), HD_, 0, stream>>>(kproj, knw, pos, Ktb, NKV_, 1);
    vpack_kernel<<<(S_ * NKV_ * HD_ + 255) / 256, 256, 0, stream>>>(vproj, Vb);

    // attention
    attn_kernel<<<dim3(S_ / 16, NH_), 32, 0, stream>>>(Qb, Ktb, Vb, attnb);

    // output projection -> f32 d_out
    gemm_bf16_kernel<<<dim3(HID_ / 64, S_ / 64), 32, 0, stream>>>(
        attnb, wob, (float*)d_out, S_, HID_, NH_ * HD_);
}
